// EncodeProcessDecode_58557584114440
// MI455X (gfx1250) — matrix operand order repaired and checked by reading, not compiled, not run on hardware
//
#include <hip/hip_runtime.h>
#include <math.h>

#define N_NODES 100000
#define N_EDGES 800000
#define L 128
#define S_STEPS 15
#define TWIN 5
#define OUTF 3

// per-step bf16 transposed weight block: ew1T[128][384] ew2T[128][128] nw1T[128][256] nw2T[128][128]
#define WSTEP 114688
#define OFF_EW2 49152
#define OFF_NW1 65536
#define OFF_NW2 98304

typedef __attribute__((ext_vector_type(16))) __bf16 v16bf;
typedef __attribute__((ext_vector_type(8)))  float  v8f;

__device__ __forceinline__ unsigned short f2bf(float f) {
    unsigned int u = __float_as_uint(f);
    return (unsigned short)((u + 0x7FFFu + ((u >> 16) & 1u)) >> 16);
}
__device__ __forceinline__ __bf16 us2bf(unsigned short u) {
    union { unsigned short u; __bf16 b; } cv; cv.u = u; return cv.b;
}

// A fragment: 16 rows x 32 K slice of row-major [16][ktot] bf16 tile (in LDS).
// A 16x32 layout (matches B): lane&15 = row; lanes 0-15 hold K 0..15, lanes 16-31 K 16..31.
__device__ __forceinline__ v16bf load_a_frag(const unsigned short* Ab, int ktot, int k0, int lane) {
    const int row   = lane & 15;
    const int khalf = (lane >> 4) & 1;
    const unsigned short* p = Ab + row * ktot + k0 + khalf * 16;
    v16bf a;
#pragma unroll
    for (int e = 0; e < 8; ++e) a[e] = us2bf(p[e]);
#pragma unroll
    for (int e = 0; e < 8; ++e) a[8 + e] = us2bf(p[8 + e]);
    return a;
}

// B fragment: 32 K x 16 cols from transposed weights wT[col][ktot] (global, L2-resident).
// B layout used here (same k order as A): lane&15 = col; lanes 0-15 hold K 0..15, lanes 16-31 K 16..31 (contiguous -> b128 loads).
__device__ __forceinline__ v16bf load_b_frag(const unsigned short* wT, int ktot, int k0, int col0, int lane) {
    const int col = col0 + (lane & 15);
    const unsigned short* p = wT + (size_t)col * ktot + k0 + ((lane >> 4) & 1) * 16;
    v16bf b;
#pragma unroll
    for (int e = 0; e < 16; ++e) b[e] = us2bf(p[e]);
    return b;
}

// ---------------- weight prep: f32 [K][128] -> bf16 transposed [128][K] ----------------
__global__ void prep_weights_kernel(const float* __restrict__ ew1, const float* __restrict__ ew2,
                                    const float* __restrict__ nw1, const float* __restrict__ nw2,
                                    unsigned short* __restrict__ wbf) {
    const size_t total = (size_t)S_STEPS * WSTEP;
    for (size_t i = (size_t)blockIdx.x * blockDim.x + threadIdx.x; i < total;
         i += (size_t)gridDim.x * blockDim.x) {
        int s = (int)(i / WSTEP);
        int r = (int)(i % WSTEP);
        float v;
        if (r < OFF_EW2) {                 // ew1T [128][384]
            int n = r / 384, k = r % 384;
            v = ew1[(size_t)s * 49152 + (size_t)k * 128 + n];
        } else if (r < OFF_NW1) {          // ew2T [128][128]
            int q = r - OFF_EW2; int n = q >> 7, k = q & 127;
            v = ew2[(size_t)s * 16384 + (size_t)k * 128 + n];
        } else if (r < OFF_NW2) {          // nw1T [128][256]
            int q = r - OFF_NW1; int n = q >> 8, k = q & 255;
            v = nw1[(size_t)s * 32768 + (size_t)k * 128 + n];
        } else {                           // nw2T [128][128]
            int q = r - OFF_NW2; int n = q >> 7, k = q & 127;
            v = nw2[(size_t)s * 16384 + (size_t)k * 128 + n];
        }
        wbf[i] = f2bf(v);
    }
}

__global__ void zero_kernel(float* __restrict__ p, size_t n) {
    for (size_t i = (size_t)blockIdx.x * blockDim.x + threadIdx.x; i < n;
         i += (size_t)gridDim.x * blockDim.x) p[i] = 0.0f;
}

// ---------------- encoders: small-K MLP + LayerNorm, one row per block ----------------
__global__ void __launch_bounds__(128)
node_encoder_kernel(const float* __restrict__ pvf, const float* __restrict__ mD,
                    const float* __restrict__ mX, const int* __restrict__ ntype,
                    const float* __restrict__ nmean, const float* __restrict__ nstd,
                    const float* __restrict__ w1, const float* __restrict__ b1,
                    const float* __restrict__ w2, const float* __restrict__ b2,
                    const float* __restrict__ g, const float* __restrict__ be,
                    float* __restrict__ nl, unsigned short* __restrict__ nlb) {
    __shared__ float F[8];
    __shared__ float H1[L];
    __shared__ float red[L], red2[L];
    const int n = blockIdx.x, t = threadIdx.x;
    if (t == 0) {
        int ty = ntype[n];
        float f0 = pvf[n], f1 = mD[n], f2 = mX[n];
        float f3 = (ty == 0) ? 1.f : 0.f, f4 = (ty == 1) ? 1.f : 0.f, f5 = (ty == 2) ? 1.f : 0.f;
        F[0] = (f0 - nmean[0]) / nstd[0]; F[1] = (f1 - nmean[1]) / nstd[1];
        F[2] = (f2 - nmean[2]) / nstd[2]; F[3] = (f3 - nmean[3]) / nstd[3];
        F[4] = (f4 - nmean[4]) / nstd[4]; F[5] = (f5 - nmean[5]) / nstd[5];
    }
    __syncthreads();
    float h = b1[t];
#pragma unroll
    for (int k = 0; k < 6; ++k) h += F[k] * w1[k * L + t];
    h = h > 0.f ? h : 0.f;
    H1[t] = h;
    __syncthreads();
    float h2 = b2[t];
    for (int k = 0; k < L; ++k) h2 += H1[k] * w2[k * L + t];
    h2 = h2 > 0.f ? h2 : 0.f;
    red[t] = h2; red2[t] = h2 * h2;
    __syncthreads();
    for (int off = 64; off > 0; off >>= 1) {
        if (t < off) { red[t] += red[t + off]; red2[t] += red2[t + off]; }
        __syncthreads();
    }
    float mu = red[0] * (1.f / L);
    float var = red2[0] * (1.f / L) - mu * mu;
    float val = (h2 - mu) * rsqrtf(var + 1e-5f) * g[t] + be[t];
    size_t idx = (size_t)n * L + t;
    nl[idx] = val; nlb[idx] = f2bf(val);
}

__global__ void __launch_bounds__(128)
edge_encoder_kernel(const float* __restrict__ pvf, const float* __restrict__ mpos,
                    const float* __restrict__ wpos, const int* __restrict__ snd,
                    const int* __restrict__ rcv,
                    const float* __restrict__ emean, const float* __restrict__ estd,
                    const float* __restrict__ w1, const float* __restrict__ b1,
                    const float* __restrict__ w2, const float* __restrict__ b2,
                    const float* __restrict__ g, const float* __restrict__ be,
                    float* __restrict__ el, unsigned short* __restrict__ elb) {
    __shared__ float F[8];
    __shared__ float H1[L];
    __shared__ float red[L], red2[L];
    const int e = blockIdx.x, t = threadIdx.x;
    if (t == 0) {
        int s = snd[e], r = rcv[e];
        float rmx = mpos[2 * s] - mpos[2 * r], rmy = mpos[2 * s + 1] - mpos[2 * r + 1];
        float rwx = wpos[2 * s] - wpos[2 * r], rwy = wpos[2 * s + 1] - wpos[2 * r + 1];
        float f[7] = { rmx, rmy, sqrtf(rmx * rmx + rmy * rmy),
                       rwx, rwy, sqrtf(rwx * rwx + rwy * rwy),
                       pvf[s] - pvf[r] };
#pragma unroll
        for (int k = 0; k < 7; ++k) F[k] = (f[k] - emean[k]) / estd[k];
    }
    __syncthreads();
    float h = b1[t];
#pragma unroll
    for (int k = 0; k < 7; ++k) h += F[k] * w1[k * L + t];
    h = h > 0.f ? h : 0.f;
    H1[t] = h;
    __syncthreads();
    float h2 = b2[t];
    for (int k = 0; k < L; ++k) h2 += H1[k] * w2[k * L + t];
    h2 = h2 > 0.f ? h2 : 0.f;
    red[t] = h2; red2[t] = h2 * h2;
    __syncthreads();
    for (int off = 64; off > 0; off >>= 1) {
        if (t < off) { red[t] += red[t + off]; red2[t] += red2[t + off]; }
        __syncthreads();
    }
    float mu = red[0] * (1.f / L);
    float var = red2[0] * (1.f / L) - mu * mu;
    float val = (h2 - mu) * rsqrtf(var + 1e-5f) * g[t] + be[t];
    size_t idx = (size_t)e * L + t;
    el[idx] = val; elb[idx] = f2bf(val);
}

// ---------------- edge GraphNetBlock: WMMA MLP(3L->L->L) + LN + residual + scatter-add ----------------
__global__ void __launch_bounds__(256)
edge_block_kernel(const unsigned short* __restrict__ nlb, unsigned short* __restrict__ elb,
                  float* __restrict__ el, float* __restrict__ aggr,
                  const int* __restrict__ gsnd, const int* __restrict__ grcv,
                  const unsigned short* __restrict__ wbf,
                  const float* __restrict__ eb1, const float* __restrict__ eb2,
                  const float* __restrict__ eg, const float* __restrict__ ebt, int s) {
    __shared__ unsigned short At[16][384];
    __shared__ unsigned short H1[16][128];
    __shared__ float H2[16][128];
    __shared__ float mu[16], rs[16];
    __shared__ int snd[16], rcv[16];

    const int tid = threadIdx.x;
    const int e0 = blockIdx.x * 16;
    const unsigned short* ew1T = wbf + (size_t)s * WSTEP;
    const unsigned short* ew2T = ew1T + OFF_EW2;
    const float* b1 = eb1 + s * L;
    const float* b2 = eb2 + s * L;
    const float* gv = eg + s * L;
    const float* bt = ebt + s * L;
    __builtin_prefetch(ew1T, 0, 0);

    if (tid < 16) { snd[tid] = gsnd[e0 + tid]; rcv[tid] = grcv[e0 + tid]; }
    __syncthreads();

    // stage A = [nl[s] | nl[r] | el] as bf16 in LDS
    for (int i = tid; i < 16 * 384; i += 256) {
        int row = i / 384, c = i - row * 384;
        unsigned short v;
        if (c < 128)      v = nlb[(size_t)snd[row] * L + c];
        else if (c < 256) v = nlb[(size_t)rcv[row] * L + (c - 128)];
        else              v = elb[(size_t)(e0 + row) * L + (c - 256)];
        At[row][c] = v;
    }
    __syncthreads();

    const int lane = tid & 31;
    const int col0 = (tid >> 5) * 16;

    // layer 1: [16x384] x [384x128]
    v8f acc = {0.f, 0.f, 0.f, 0.f, 0.f, 0.f, 0.f, 0.f};
#pragma unroll
    for (int kk = 0; kk < 384; kk += 32) {
        v16bf a = load_a_frag(&At[0][0], 384, kk, lane);
        v16bf b = load_b_frag(ew1T, 384, kk, col0, lane);
        acc = __builtin_amdgcn_wmma_f32_16x16x32_bf16(false, a, false, b, (short)0, acc, false, false);
    }
    {
        const int col = col0 + (lane & 15);
        const int rbase = ((lane >> 4) & 1) * 8;
        const float bias = b1[col];
#pragma unroll
        for (int v = 0; v < 8; ++v) {
            float h = acc[v] + bias;
            H1[rbase + v][col] = f2bf(h > 0.f ? h : 0.f);
        }
    }
    __syncthreads();

    // layer 2: [16x128] x [128x128]
    v8f acc2 = {0.f, 0.f, 0.f, 0.f, 0.f, 0.f, 0.f, 0.f};
#pragma unroll
    for (int kk = 0; kk < 128; kk += 32) {
        v16bf a = load_a_frag(&H1[0][0], 128, kk, lane);
        v16bf b = load_b_frag(ew2T, 128, kk, col0, lane);
        acc2 = __builtin_amdgcn_wmma_f32_16x16x32_bf16(false, a, false, b, (short)0, acc2, false, false);
    }
    {
        const int col = col0 + (lane & 15);
        const int rbase = ((lane >> 4) & 1) * 8;
        const float bias = b2[col];
#pragma unroll
        for (int v = 0; v < 8; ++v) {
            float h = acc2[v] + bias;
            H2[rbase + v][col] = h > 0.f ? h : 0.f;
        }
    }
    __syncthreads();

    if (tid < 16) {
        float s1 = 0.f, s2 = 0.f;
        for (int c = 0; c < 128; ++c) { float v = H2[tid][c]; s1 += v; s2 += v * v; }
        float m = s1 * (1.f / L);
        mu[tid] = m;
        rs[tid] = rsqrtf(s2 * (1.f / L) - m * m + 1e-5f);
    }
    __syncthreads();

    for (int i = tid; i < 16 * 128; i += 256) {
        int row = i >> 7, c = i & 127;
        float ne = (H2[row][c] - mu[row]) * rs[row] * gv[c] + bt[c];
        size_t idx = (size_t)(e0 + row) * L + c;
        float nv = el[idx] + ne;          // residual into edge latent
        el[idx] = nv;
        elb[idx] = f2bf(nv);
        atomicAdd(&aggr[(size_t)rcv[row] * L + c], ne);  // segment_sum(new_e, receivers)
    }
}

// ---------------- node GraphNetBlock: WMMA MLP(2L->L->L) + LN + residual ----------------
__global__ void __launch_bounds__(256)
node_block_kernel(float* __restrict__ nl, unsigned short* __restrict__ nlb,
                  const float* __restrict__ aggr,
                  const unsigned short* __restrict__ wbf,
                  const float* __restrict__ nb1, const float* __restrict__ nb2,
                  const float* __restrict__ ng, const float* __restrict__ nbt, int s) {
    __shared__ unsigned short At[16][256];
    __shared__ unsigned short H1[16][128];
    __shared__ float H2[16][128];
    __shared__ float mu[16], rs[16];

    const int tid = threadIdx.x;
    const int n0 = blockIdx.x * 16;
    const unsigned short* nw1T = wbf + (size_t)s * WSTEP + OFF_NW1;
    const unsigned short* nw2T = wbf + (size_t)s * WSTEP + OFF_NW2;
    const float* b1 = nb1 + s * L;
    const float* b2 = nb2 + s * L;
    const float* gv = ng + s * L;
    const float* bt = nbt + s * L;
    __builtin_prefetch(nw1T, 0, 0);

    for (int i = tid; i < 16 * 256; i += 256) {
        int row = i >> 8, c = i & 255;
        size_t base = (size_t)(n0 + row) * L;
        At[row][c] = (c < 128) ? nlb[base + c] : f2bf(aggr[base + (c - 128)]);
    }
    __syncthreads();

    const int lane = tid & 31;
    const int col0 = (tid >> 5) * 16;

    v8f acc = {0.f, 0.f, 0.f, 0.f, 0.f, 0.f, 0.f, 0.f};
#pragma unroll
    for (int kk = 0; kk < 256; kk += 32) {
        v16bf a = load_a_frag(&At[0][0], 256, kk, lane);
        v16bf b = load_b_frag(nw1T, 256, kk, col0, lane);
        acc = __builtin_amdgcn_wmma_f32_16x16x32_bf16(false, a, false, b, (short)0, acc, false, false);
    }
    {
        const int col = col0 + (lane & 15);
        const int rbase = ((lane >> 4) & 1) * 8;
        const float bias = b1[col];
#pragma unroll
        for (int v = 0; v < 8; ++v) {
            float h = acc[v] + bias;
            H1[rbase + v][col] = f2bf(h > 0.f ? h : 0.f);
        }
    }
    __syncthreads();

    v8f acc2 = {0.f, 0.f, 0.f, 0.f, 0.f, 0.f, 0.f, 0.f};
#pragma unroll
    for (int kk = 0; kk < 128; kk += 32) {
        v16bf a = load_a_frag(&H1[0][0], 128, kk, lane);
        v16bf b = load_b_frag(nw2T, 128, kk, col0, lane);
        acc2 = __builtin_amdgcn_wmma_f32_16x16x32_bf16(false, a, false, b, (short)0, acc2, false, false);
    }
    {
        const int col = col0 + (lane & 15);
        const int rbase = ((lane >> 4) & 1) * 8;
        const float bias = b2[col];
#pragma unroll
        for (int v = 0; v < 8; ++v) {
            float h = acc2[v] + bias;
            H2[rbase + v][col] = h > 0.f ? h : 0.f;
        }
    }
    __syncthreads();

    if (tid < 16) {
        float s1 = 0.f, s2 = 0.f;
        for (int c = 0; c < 128; ++c) { float v = H2[tid][c]; s1 += v; s2 += v * v; }
        float m = s1 * (1.f / L);
        mu[tid] = m;
        rs[tid] = rsqrtf(s2 * (1.f / L) - m * m + 1e-5f);
    }
    __syncthreads();

    for (int i = tid; i < 16 * 128; i += 256) {
        int row = i >> 7, c = i & 127;
        float nn = (H2[row][c] - mu[row]) * rs[row] * gv[c] + bt[c];
        size_t idx = (size_t)(n0 + row) * L + c;
        float nv = nl[idx] + nn;          // residual into node latent
        nl[idx] = nv;
        nlb[idx] = f2bf(nv);
    }
}

// ---------------- decoder: 128 -> 8 (swish) -> 15, scale by dt, transpose to [TW,N,OUT] ----------------
__global__ void __launch_bounds__(256)
decode_kernel(const float* __restrict__ nl, const float* __restrict__ w1,
              const float* __restrict__ b1, const float* __restrict__ w2,
              const float* __restrict__ b2, float* __restrict__ out) {
    const int n = blockIdx.x * blockDim.x + threadIdx.x;
    if (n >= N_NODES) return;
    float h[8];
#pragma unroll
    for (int j = 0; j < 8; ++j) h[j] = b1[j];
    const float* x = nl + (size_t)n * L;
    for (int k = 0; k < L; ++k) {
        float xv = x[k];
#pragma unroll
        for (int j = 0; j < 8; ++j) h[j] += xv * w1[k * 8 + j];
    }
#pragma unroll
    for (int j = 0; j < 8; ++j) { float v = h[j]; h[j] = v / (1.f + expf(-v)); }  // swish
#pragma unroll
    for (int m = 0; m < OUTF * TWIN; ++m) {
        float o = b2[m];
#pragma unroll
        for (int j = 0; j < 8; ++j) o += h[j] * w2[j * (OUTF * TWIN) + m];
        int tw = m / OUTF, c = m - tw * OUTF;
        out[((size_t)tw * N_NODES + n) * OUTF + c] = o * (float)(tw + 1);
    }
}

extern "C" void kernel_launch(void* const* d_in, const int* in_sizes, int n_in,
                              void* d_out, int out_size, void* d_ws, size_t ws_size,
                              hipStream_t stream) {
    const float* pvf   = (const float*)d_in[0];
    const float* matD  = (const float*)d_in[1];
    const float* matX  = (const float*)d_in[2];
    const int*   ntype = (const int*)d_in[3];
    const float* mpos  = (const float*)d_in[4];
    const float* wpos  = (const float*)d_in[5];
    const int*   snd   = (const int*)d_in[6];
    const int*   rcv   = (const int*)d_in[7];
    const float* nmean = (const float*)d_in[8];
    const float* nstd  = (const float*)d_in[9];
    const float* emean = (const float*)d_in[10];
    const float* estd  = (const float*)d_in[11];
    const float* ne_w1 = (const float*)d_in[12];
    const float* ne_b1 = (const float*)d_in[13];
    const float* ne_w2 = (const float*)d_in[14];
    const float* ne_b2 = (const float*)d_in[15];
    const float* ne_g  = (const float*)d_in[16];
    const float* ne_be = (const float*)d_in[17];
    const float* ee_w1 = (const float*)d_in[18];
    const float* ee_b1 = (const float*)d_in[19];
    const float* ee_w2 = (const float*)d_in[20];
    const float* ee_b2 = (const float*)d_in[21];
    const float* ee_g  = (const float*)d_in[22];
    const float* ee_be = (const float*)d_in[23];
    const float* blk_ew1 = (const float*)d_in[24];
    const float* blk_eb1 = (const float*)d_in[25];
    const float* blk_ew2 = (const float*)d_in[26];
    const float* blk_eb2 = (const float*)d_in[27];
    const float* blk_eg  = (const float*)d_in[28];
    const float* blk_ebt = (const float*)d_in[29];
    const float* blk_nw1 = (const float*)d_in[30];
    const float* blk_nb1 = (const float*)d_in[31];
    const float* blk_nw2 = (const float*)d_in[32];
    const float* blk_nb2 = (const float*)d_in[33];
    const float* blk_ng  = (const float*)d_in[34];
    const float* blk_nbt = (const float*)d_in[35];
    const float* dec_w1 = (const float*)d_in[36];
    const float* dec_b1 = (const float*)d_in[37];
    const float* dec_w2 = (const float*)d_in[38];
    const float* dec_b2 = (const float*)d_in[39];
    float* out = (float*)d_out;

    // workspace carve-up
    char* w = (char*)d_ws;
    float* nl = (float*)w;              w += sizeof(float) * (size_t)N_NODES * L;
    float* el = (float*)w;              w += sizeof(float) * (size_t)N_EDGES * L;
    float* aggr = (float*)w;            w += sizeof(float) * (size_t)N_NODES * L;
    unsigned short* nlb = (unsigned short*)w;  w += sizeof(unsigned short) * (size_t)N_NODES * L;
    unsigned short* elb = (unsigned short*)w;  w += sizeof(unsigned short) * (size_t)N_EDGES * L;
    unsigned short* wbf = (unsigned short*)w;  w += sizeof(unsigned short) * (size_t)S_STEPS * WSTEP;

    prep_weights_kernel<<<1024, 256, 0, stream>>>(blk_ew1, blk_ew2, blk_nw1, blk_nw2, wbf);
    node_encoder_kernel<<<N_NODES, 128, 0, stream>>>(pvf, matD, matX, ntype, nmean, nstd,
                                                     ne_w1, ne_b1, ne_w2, ne_b2, ne_g, ne_be, nl, nlb);
    edge_encoder_kernel<<<N_EDGES, 128, 0, stream>>>(pvf, mpos, wpos, snd, rcv, emean, estd,
                                                     ee_w1, ee_b1, ee_w2, ee_b2, ee_g, ee_be, el, elb);
    for (int s = 0; s < S_STEPS; ++s) {
        zero_kernel<<<2048, 256, 0, stream>>>(aggr, (size_t)N_NODES * L);
        edge_block_kernel<<<N_EDGES / 16, 256, 0, stream>>>(nlb, elb, el, aggr, snd, rcv, wbf,
                                                            blk_eb1, blk_eb2, blk_eg, blk_ebt, s);
        node_block_kernel<<<N_NODES / 16, 256, 0, stream>>>(nl, nlb, aggr, wbf,
                                                            blk_nb1, blk_nb2, blk_ng, blk_nbt, s);
    }
    decode_kernel<<<(N_NODES + 255) / 256, 256, 0, stream>>>(nl, dec_w1, dec_b1, dec_w2, dec_b2, out);
}
